// CLIPVisionTower_7876970021578
// MI455X (gfx1250) — matrix operand order repaired and checked by reading, not compiled, not run on hardware
//
#include <hip/hip_runtime.h>
#include <hip/hip_bf16.h>

// ---------------------------------------------------------------------------
// CLIP vision-tower token pruning/merging (EViT-style) for MI455X / gfx1250.
//
// Shapes: image_features (32,576,1024) f32; q,k (32,577,1024) f32.
// Output: (32,73,1024) f32.
//
// Key insight: only attention row 0 is needed (cls_attn), and every gather in
// the reference is a permutation perm = [top72_idx, complement]. The only
// heavy math is sims = NK[0:72] @ NK[0:576]^T per batch (~2.7 GFLOP total),
// done here with v_wmma_f32_16x16x32_bf16 (normalized keys -> bf16 is ample
// precision for a similarity ranking, and halves matmul bytes).
// ---------------------------------------------------------------------------

typedef __attribute__((ext_vector_type(16))) __bf16 v16bf;
typedef __attribute__((ext_vector_type(8)))  float  v8f;

#define BB   32
#define NN   576
#define CC   1024
#define NQ   577
#define KK   72      // top-k kept tokens
#define CK   32      // cluster size
#define NEG_INF (-__builtin_inff())

// ---- workspace byte offsets (d_ws), 256B aligned -------------------------
#define OFF_LOGITS 0u                                   // 32*577 f32
#define OFF_ATTN   73984u                               // 32*576 f32
#define OFF_PERM   147712u                              // 32*576 i32
#define OFF_PATTN  221440u                              // 32*576 f32 (attn in perm order)
#define OFF_PK     295168u                              // 32*576*1024 bf16 (normalized keys, perm order)
#define OFF_SIMS   38043904u                            // 32*72*576 f32

__device__ __forceinline__ unsigned short f2bf_rne(float f) {
    unsigned int u = __float_as_uint(f);
    unsigned int r = u + 0x7FFFu + ((u >> 16) & 1u);
    return (unsigned short)(r >> 16);
}

// ---------------------------------------------------------------------------
// K1: cls logits: logits[b][j] = (1/32) * dot(q[b,0,:], k[b,j,:]), one wave/j
// ---------------------------------------------------------------------------
__global__ void k1_cls_logits(const float* __restrict__ q,
                              const float* __restrict__ k,
                              float* __restrict__ logits) {
    int wave = blockIdx.x * 8 + (threadIdx.x >> 5);   // 32*577 waves total
    int lane = threadIdx.x & 31;
    int b = wave / NQ;
    int j = wave - b * NQ;
    const float* qr = q + (size_t)b * NQ * CC;            // row 0
    const float* kr = k + ((size_t)b * NQ + j) * CC;
    float s = 0.f;
    #pragma unroll
    for (int c = lane; c < CC; c += 32) s += qr[c] * kr[c];
    #pragma unroll
    for (int off = 16; off > 0; off >>= 1) s += __shfl_xor(s, off, 32);
    if (lane == 0) logits[b * NQ + j] = s * 0.03125f;   // C^-0.5 = 1/32
}

// ---------------------------------------------------------------------------
// K2: per-batch softmax over 577 logits; write cls_attn[b][0..575] (= [1:])
// ---------------------------------------------------------------------------
__global__ void k2_softmax(const float* __restrict__ logits,
                           float* __restrict__ attn) {
    __shared__ float v[1024], red[1024];
    int tid = threadIdx.x, b = blockIdx.x;
    float x = (tid < NQ) ? logits[b * NQ + tid] : NEG_INF;
    v[tid] = x; red[tid] = x;
    __syncthreads();
    for (int s = 512; s > 0; s >>= 1) {
        if (tid < s) red[tid] = fmaxf(red[tid], red[tid + s]);
        __syncthreads();
    }
    float m = red[0];
    __syncthreads();
    float e = (tid < NQ) ? __expf(v[tid] - m) : 0.f;
    red[tid] = e;
    __syncthreads();
    for (int s = 512; s > 0; s >>= 1) {
        if (tid < s) red[tid] += red[tid + s];
        __syncthreads();
    }
    float inv = 1.f / red[0];
    if (tid >= 1 && tid < NQ) attn[b * NN + (tid - 1)] = e * inv;
}

// ---------------------------------------------------------------------------
// K3: per-batch top-72 (iterative argmax) + complement via prefix sum
//     -> perm[b][0..575], pattn[b][j] = attn[b][perm[j]]
// ---------------------------------------------------------------------------
__global__ void k3_topk_perm(const float* __restrict__ attn,
                             int* __restrict__ perm,
                             float* __restrict__ pattn) {
    __shared__ float v[1024], orig[NN], rv[1024];
    __shared__ int   rj[1024], sc[1024], sel[NN], lperm[NN];
    int tid = threadIdx.x, b = blockIdx.x;
    float a = (tid < NN) ? attn[b * NN + tid] : NEG_INF;
    v[tid] = a;
    if (tid < NN) { orig[tid] = a; sel[tid] = 0; }
    __syncthreads();
    // 72 argmax rounds
    for (int r = 0; r < KK; ++r) {
        rv[tid] = v[tid]; rj[tid] = tid;
        __syncthreads();
        for (int s = 512; s > 0; s >>= 1) {
            if (tid < s && rv[tid + s] > rv[tid]) { rv[tid] = rv[tid + s]; rj[tid] = rj[tid + s]; }
            __syncthreads();
        }
        if (tid == 0) { int am = rj[0]; lperm[r] = am; v[am] = NEG_INF; sel[am] = 1; }
        __syncthreads();
    }
    // complement in ascending index order via inclusive Hillis-Steele scan
    int flag = (tid < NN && !sel[tid]) ? 1 : 0;
    sc[tid] = flag;
    __syncthreads();
    for (int off = 1; off < 1024; off <<= 1) {
        int t = (tid >= off) ? sc[tid - off] : 0;
        __syncthreads();
        sc[tid] += t;
        __syncthreads();
    }
    if (flag) lperm[KK + sc[tid] - 1] = tid;
    __syncthreads();
    if (tid < NN) {
        int p = lperm[tid];
        perm[b * NN + tid] = p;
        pattn[b * NN + tid] = orig[p];
    }
}

// ---------------------------------------------------------------------------
// K4: L2-normalize key rows (excluding cls) in PERMUTED order -> bf16 pk
//     one block per (b, j): row = perm[b][j], source k[b, row+1, :]
// ---------------------------------------------------------------------------
__global__ void k4_norm_keys(const float* __restrict__ k,
                             const int* __restrict__ perm,
                             unsigned short* __restrict__ pk) {
    __shared__ float red[256];
    int tid = threadIdx.x;
    int j = blockIdx.x % NN;
    int b = blockIdx.x / NN;
    int row = perm[b * NN + j];
    const float* kr = k + ((size_t)b * NQ + row + 1) * CC;
    float x0 = kr[tid], x1 = kr[tid + 256], x2 = kr[tid + 512], x3 = kr[tid + 768];
    red[tid] = x0 * x0 + x1 * x1 + x2 * x2 + x3 * x3;
    __syncthreads();
    for (int s = 128; s > 0; s >>= 1) {
        if (tid < s) red[tid] += red[tid + s];
        __syncthreads();
    }
    float inv = 1.f / fmaxf(sqrtf(red[0]), 1e-12f);
    unsigned short* pr = pk + ((size_t)b * NN + j) * CC;
    pr[tid]       = f2bf_rne(x0 * inv);
    pr[tid + 256] = f2bf_rne(x1 * inv);
    pr[tid + 512] = f2bf_rne(x2 * inv);
    pr[tid + 768] = f2bf_rne(x3 * inv);
}

// ---------------------------------------------------------------------------
// K5: WMMA sims = PK[0:72] @ PK[0:576]^T  per batch, f32 accum, diag = -inf.
// One wave per 16x16 output tile; K loop 1024 in steps of 32 (bf16 WMMA).
// Per-lane operand loads are contiguous uint4's per the §7.12.2 layouts:
//   A 16x32: lanes 0-15 row m, K{0..7,16..23}; lanes 16-31 row m, K{8..15,24..31}
//   B 32x16: lanes 0-15 col n, K{0..7,16..23}; lanes 16-31 col n, K{8..15,24..31}
// (B columns are rows of PK -> two 16-byte chunks per lane, 32 bytes apart.)
// ---------------------------------------------------------------------------
#define MT 5      // ceil(72/16)
#define NT 36     // 576/16
union PackBF { uint4 u[2]; v16bf v; };

__global__ void k5_wmma_sims(const unsigned short* __restrict__ pk,
                             float* __restrict__ sims) {
    int tile = blockIdx.x * 8 + (threadIdx.x >> 5);   // 32*5*36 = 5760 tiles
    int lane = threadIdx.x & 31;
    int b  = tile / (MT * NT);
    int r0 = tile - b * (MT * NT);
    int mt = r0 / NT;
    int nt = r0 - mt * NT;
    int m0 = mt * 16, n0 = nt * 16;
    int l  = lane & 15, hs = lane >> 4;

    const unsigned short* pkb  = pk + (size_t)b * NN * CC;
    const unsigned short* arow = pkb + (size_t)(m0 + l) * CC;
    const unsigned short* brow = pkb + (size_t)(n0 + l) * CC;

    v8f c = {};
    for (int k0 = 0; k0 < CC; k0 += 32) {
        PackBF A, Bm;
        A.u[0]  = *(const uint4*)(arow + k0 + hs * 8);
        A.u[1]  = *(const uint4*)(arow + k0 + 16 + hs * 8);
        const unsigned short* bp = brow + k0 + hs * 8;
        Bm.u[0] = *(const uint4*)(bp);
        Bm.u[1] = *(const uint4*)(bp + 16);
        c = __builtin_amdgcn_wmma_f32_16x16x32_bf16(
                false, A.v, false, Bm.v, (short)0, c, false, false);
    }
    // C/D layout: VGPR r -> (M = m0 + hs*8 + r, N = n0 + l)
    int col = n0 + l;
    float* sb = sims + (size_t)b * KK * NN;
    #pragma unroll
    for (int r = 0; r < 8; ++r) {
        int mrow = m0 + hs * 8 + r;
        if (mrow < KK) {
            float vv = c[r];
            if (mrow == col) vv = NEG_INF;     // mask self-similarity
            sb[(size_t)mrow * NN + col] = vv;
        }
    }
}

// ---------------------------------------------------------------------------
// K6: per (b,i<72): top-32 of sims row, then
//     out[b,i,:] = x[perm[i]] + sum_c w[cidx_c] * x[perm[cidx_c]]
// ---------------------------------------------------------------------------
__global__ void k6_cluster_sum(const float* __restrict__ sims,
                               const float* __restrict__ pattn,
                               const int* __restrict__ perm,
                               const float* __restrict__ imf,
                               float* __restrict__ out) {
    __shared__ float sv[NN], lw[NN], rmax[256], cw[CK];
    __shared__ int   lperm[NN], rjm[256], ctok[CK];
    int tid = threadIdx.x;
    int i = blockIdx.x % KK;
    int b = blockIdx.x / KK;
    const float* srow = sims + ((size_t)b * KK + i) * NN;
    for (int j = tid; j < NN; j += 256) {
        sv[j]    = srow[j];
        lw[j]    = pattn[b * NN + j];
        lperm[j] = perm[b * NN + j];
    }
    __syncthreads();
    for (int r = 0; r < CK; ++r) {
        float bv = NEG_INF; int bj = 0;
        for (int j = tid; j < NN; j += 256)
            if (sv[j] > bv) { bv = sv[j]; bj = j; }
        rmax[tid] = bv; rjm[tid] = bj;
        __syncthreads();
        for (int s = 128; s > 0; s >>= 1) {
            if (tid < s && rmax[tid + s] > rmax[tid]) { rmax[tid] = rmax[tid + s]; rjm[tid] = rjm[tid + s]; }
            __syncthreads();
        }
        if (tid == 0) { int am = rjm[0]; ctok[r] = lperm[am]; cw[r] = lw[am]; sv[am] = NEG_INF; }
        __syncthreads();
    }
    // weighted accumulation over 32 cluster tokens + self
    int ch = tid * 4;
    const float* ib = imf + (size_t)b * NN * CC;
    float4 acc = *(const float4*)(ib + (size_t)lperm[i] * CC + ch);
    #pragma unroll 4
    for (int cc = 0; cc < CK; ++cc) {
        float w = cw[cc];
        float4 t = *(const float4*)(ib + (size_t)ctok[cc] * CC + ch);
        acc.x += w * t.x; acc.y += w * t.y; acc.z += w * t.z; acc.w += w * t.w;
    }
    *(float4*)(out + ((size_t)b * (KK + 1) + i) * CC + ch) = acc;
}

// ---------------------------------------------------------------------------
// K7: extra row: out[b,72,:] = sum_{j=72..575} pattn[j] * x[perm[j]]
// ---------------------------------------------------------------------------
__global__ void k7_extra(const float* __restrict__ pattn,
                         const int* __restrict__ perm,
                         const float* __restrict__ imf,
                         float* __restrict__ out) {
    __shared__ float lw[NN];
    __shared__ int   lp[NN];
    int tid = threadIdx.x, b = blockIdx.x;
    for (int j = tid; j < NN; j += 256) {
        lw[j] = pattn[b * NN + j];
        lp[j] = perm[b * NN + j];
    }
    __syncthreads();
    int ch = tid * 4;
    const float* ib = imf + (size_t)b * NN * CC;
    float4 acc = make_float4(0.f, 0.f, 0.f, 0.f);
    for (int j = KK; j < NN; ++j) {
        float w = lw[j];
        float4 t = *(const float4*)(ib + (size_t)lp[j] * CC + ch);
        acc.x += w * t.x; acc.y += w * t.y; acc.z += w * t.z; acc.w += w * t.w;
    }
    *(float4*)(out + ((size_t)b * (KK + 1) + KK) * CC + ch) = acc;
}

// ---------------------------------------------------------------------------
extern "C" void kernel_launch(void* const* d_in, const int* in_sizes, int n_in,
                              void* d_out, int out_size, void* d_ws, size_t ws_size,
                              hipStream_t stream) {
    const float* imf = (const float*)d_in[0];
    const float* q   = (const float*)d_in[1];
    const float* k   = (const float*)d_in[2];
    float* out = (float*)d_out;

    char* ws = (char*)d_ws;
    float*          logits = (float*)(ws + OFF_LOGITS);
    float*          attn   = (float*)(ws + OFF_ATTN);
    int*            perm   = (int*)  (ws + OFF_PERM);
    float*          pattn  = (float*)(ws + OFF_PATTN);
    unsigned short* pk     = (unsigned short*)(ws + OFF_PK);
    float*          sims   = (float*)(ws + OFF_SIMS);

    k1_cls_logits<<<(BB * NQ) / 8, 256, 0, stream>>>(q, k, logits);
    k2_softmax<<<BB, 1024, 0, stream>>>(logits, attn);
    k3_topk_perm<<<BB, 1024, 0, stream>>>(attn, perm, pattn);
    k4_norm_keys<<<BB * NN, 256, 0, stream>>>(k, perm, pk);
    k5_wmma_sims<<<(BB * MT * NT) / 8, 256, 0, stream>>>(pk, sims);
    k6_cluster_sum<<<BB * KK, 256, 0, stream>>>(sims, pattn, perm, imf, out);
    k7_extra<<<BB, 256, 0, stream>>>(pattn, perm, imf, out);
}
